// Model_16140487098554
// MI455X (gfx1250) — matrix operand order repaired and checked by reading, not compiled, not run on hardware
//
#include <hip/hip_runtime.h>
#include <hip/hip_bf16.h>

#define BSZ   64
#define SEQ   2048
#define D_IN  512
#define D_OUT 1024
#define RANK  8

typedef __attribute__((ext_vector_type(16))) __bf16       v16bf;
typedef __attribute__((ext_vector_type(8)))  float        v8f;
typedef __attribute__((ext_vector_type(4)))  unsigned int v4u;

union BV {
  v16bf bf;
  v4u   q[2];
};

__device__ __forceinline__ unsigned int pack2bf(float lo, float hi) {
  __hip_bfloat16 l = __float2bfloat16(lo);
  __hip_bfloat16 h = __float2bfloat16(hi);
  unsigned short ls, hs;
  __builtin_memcpy(&ls, &l, 2);
  __builtin_memcpy(&hs, &h, 2);
  return ((unsigned int)hs << 16) | (unsigned int)ls;
}

// ---------------------------------------------------------------------------
// Phase 1: h^T = A_lora (8x512, rows padded to 16 by duplication) * x^T.
//   D tile: M = r (rows 0..7 valid), N = s (16 rows of this s-tile).
//   A-operand rows m>=8 read clamped duplicates -> garbage rows never stored,
//   so the K-loop is branch-free: 4x global_load_b128 + 1 WMMA per iter.
//   Each lane (0..15) ends up holding h[s0+lane, r=0..7] in c[0..7] ->
//   packed into one global_store_b128.
// ---------------------------------------------------------------------------
__global__ __launch_bounds__(256) void lora_h_kernel(
    const __hip_bfloat16* __restrict__ x,
    const __hip_bfloat16* __restrict__ A,
    __hip_bfloat16* __restrict__ h)
{
  const int lane = threadIdx.x & 31;
  const int wg   = blockIdx.x * 8 + (threadIdx.x >> 5);   // global wave id
  const int b    = wg >> 7;                               // 128 s-tiles / batch
  const int st   = wg & 127;
  const int s0   = st * 16;

  const __hip_bfloat16* xb = x + ((size_t)b * SEQ + s0) * D_IN;
  const __hip_bfloat16* Ab = A + (size_t)b * RANK * D_IN;

  const int half = lane >> 4;     // 0 or 1
  const int l15  = lane & 15;
  const int mc   = lane & 7;      // clamped rank row (rows 8..15 duplicate)
  const int kha  = half << 3;     // A-operand K sub-offset: 0 or 8
  const int khb  = half << 4;     // unused; the B-operand also uses kha

  const __hip_bfloat16* arow = Ab + (size_t)mc  * D_IN;   // LoRA-A row
  const __hip_bfloat16* xrow = xb + (size_t)l15 * D_IN;   // x row (B-operand col)

  v8f c = {};
  #pragma unroll 4
  for (int k0 = 0; k0 < D_IN; k0 += 32) {
    BV a, bm;
    a.q[0]  = *(const v4u*)(arow + k0 + kha);        // K base .. base+7
    a.q[1]  = *(const v4u*)(arow + k0 + 16 + kha);   // K base+16 .. base+23
    bm.q[0] = *(const v4u*)(xrow + k0 + kha);        // K base .. base+7
    bm.q[1] = *(const v4u*)(xrow + k0 + 16 + kha);   // K base+16 .. base+23
    c = __builtin_amdgcn_wmma_f32_16x16x32_bf16(false, a.bf, false, bm.bf,
                                                (short)0, c, false, false);
  }

  // Lanes 0..15 hold h[s0+lane, r=0..7]; rows M=8..15 (lanes 16..31) are junk.
  if (lane < 16) {
    v4u st16;
    st16[0] = pack2bf(c[0], c[1]);
    st16[1] = pack2bf(c[2], c[3]);
    st16[2] = pack2bf(c[4], c[5]);
    st16[3] = pack2bf(c[6], c[7]);
    *(v4u*)(h + ((size_t)b * SEQ + s0 + lane) * RANK) = st16;
  }
}

// ---------------------------------------------------------------------------
// Phase 2: delta^T tile = B (16x8, K padded to 32) * h^T (8x16 padded).
//   B-operand (h^T) is loop-invariant across the 16 o-tiles -> hoisted.
//   Per o-tile: 1 clamped b128 load + v_cndmask zeroing + 1 WMMA + 1 b128
//   store. Each lane stores delta[s0 + (lane&15), o0 + 8*(lane>>4) .. +7]
//   = 16 contiguous bytes; one store instruction covers the whole 16x16 tile.
// ---------------------------------------------------------------------------
__global__ __launch_bounds__(256) void lora_delta_kernel(
    const __hip_bfloat16* __restrict__ h,
    const __hip_bfloat16* __restrict__ Bm,
    __hip_bfloat16* __restrict__ out)
{
  const int lane = threadIdx.x & 31;
  const int wg   = blockIdx.x * 8 + (threadIdx.x >> 5);
  const int b    = wg >> 9;            // 512 waves per batch (128 st * 4 og)
  const int rem  = wg & 511;
  const int st   = rem >> 2;
  const int og   = rem & 3;
  const int s0   = st * 16;

  const int  half = lane >> 4;
  const int  l15  = lane & 15;
  const bool lo   = (half == 0);
  const v4u  zero = (v4u)(0u);

  // B-operand (h^T, 32x16): lanes 0-15 hold K=0..15 of column s = s0+l15;
  // only K 0..7 (the real rank) nonzero. Lanes 16-31 (K 16..31) all zero.
  const v4u ht = *(const v4u*)(h + ((size_t)b * SEQ + s0 + l15) * RANK);
  BV bm;
  bm.q[0] = lo ? ht : zero;
  bm.q[1] = zero;

  const __hip_bfloat16* Bb   = Bm  + (size_t)b * D_OUT * RANK;
  __hip_bfloat16*       outb = out + ((size_t)b * SEQ + s0) * D_OUT;

  for (int ot = 0; ot < 16; ++ot) {
    const int o0 = og * 256 + ot * 16;

    // A-operand (B rows, 16x32): lane row o = o0+l15, K 0..7 valid (16 B),
    // all other K positions zero (rank padding). Clamped load + select,
    // no exec divergence.
    const v4u bt = *(const v4u*)(Bb + (size_t)(o0 + l15) * RANK);
    BV a;
    a.q[0] = lo ? bt : zero;
    a.q[1] = zero;

    v8f c = {};
    c = __builtin_amdgcn_wmma_f32_16x16x32_bf16(false, a.bf, false, bm.bf,
                                                (short)0, c, false, false);

    // c[v] = delta[s0+l15, o0 + 8*half + v], v = 0..7 -> 16 contiguous bytes.
    v4u st16;
    st16[0] = pack2bf(c[0], c[1]);
    st16[1] = pack2bf(c[2], c[3]);
    st16[2] = pack2bf(c[4], c[5]);
    st16[3] = pack2bf(c[6], c[7]);
    *(v4u*)(outb + (size_t)l15 * D_OUT + o0 + (half << 3)) = st16;
  }
}

extern "C" void kernel_launch(void* const* d_in, const int* in_sizes, int n_in,
                              void* d_out, int out_size, void* d_ws, size_t ws_size,
                              hipStream_t stream) {
  const __hip_bfloat16* x  = (const __hip_bfloat16*)d_in[0];
  const __hip_bfloat16* A  = (const __hip_bfloat16*)d_in[1];
  const __hip_bfloat16* B  = (const __hip_bfloat16*)d_in[2];
  __hip_bfloat16* out = (__hip_bfloat16*)d_out;
  __hip_bfloat16* hws = (__hip_bfloat16*)d_ws;   // 64*2048*8 bf16 = 2 MB scratch

  // Phase 1: 64 batches * 128 s-tiles = 8192 waves, 8 waves/block.
  lora_h_kernel<<<(BSZ * (SEQ / 16)) / 8, 256, 0, stream>>>(x, A, hws);
  // Phase 2: 8192 s-tiles * 4 o-groups = 32768 waves, 8 waves/block.
  lora_delta_kernel<<<(BSZ * (SEQ / 16) * 4) / 8, 256, 0, stream>>>(hws, B, out);
}
